// ImprovedEpisodicMemory_83846351553128
// MI455X (gfx1250) — matrix operand order repaired and checked by reading, not compiled, not run on hardware
//
#include <hip/hip_runtime.h>

#define N_Q   8192
#define M_MEM 8192
#define H_DIM 5120
#define P_DIM 1280
#define E_DIM 128

typedef __attribute__((ext_vector_type(16))) _Float16 v16h;
typedef __attribute__((ext_vector_type(8)))  float    v8f;

union FragH {
    v16h     v;
    _Float16 h[16];
    uint4    q[2];
};

union Pack8 {
    _Float16 h[8];
    uint4    q;
};

static __device__ __forceinline__ v8f wmma_f16(const FragH& a, const FragH& b, v8f c) {
    // D = A(16x32 f16) * B(32x16 f16) + C(16x16 f32)
    return __builtin_amdgcn_wmma_f32_16x16x32_f16(
        /*neg_a=*/false, a.v, /*neg_b=*/false, b.v,
        /*c_mod=*/(short)0, c, /*reuse_a=*/false, /*reuse_b=*/false);
}

__device__ __forceinline__ void ins3(float d, int idx, float (&dd)[3], int (&ii)[3]) {
    if (d < dd[2]) {
        if (d < dd[1]) {
            dd[2] = dd[1]; ii[2] = ii[1];
            if (d < dd[0]) { dd[1] = dd[0]; ii[1] = ii[0]; dd[0] = d; ii[0] = idx; }
            else           { dd[1] = d;     ii[1] = idx; }
        } else { dd[2] = d; ii[2] = idx; }
    }
}

__device__ __forceinline__ void cvt8(const float4& a, const float4& b, Pack8& p) {
    p.h[0]=(_Float16)a.x; p.h[1]=(_Float16)a.y; p.h[2]=(_Float16)a.z; p.h[3]=(_Float16)a.w;
    p.h[4]=(_Float16)b.x; p.h[5]=(_Float16)b.y; p.h[6]=(_Float16)b.z; p.h[7]=(_Float16)b.w;
}

// --------------------------------------------------------------------------
// C[rows x P] = A[rows x H] * W[P x H]^T (NT GEMM), f16 output.
// Block tile 128x64, 8 waves in 4x2 grid, each wave 2x2 WMMA tiles (4/step).
// Double-buffered LDS: one barrier per K-step, global loads overlap WMMA.
// --------------------------------------------------------------------------
__global__ __launch_bounds__(256) void proj_kernel(const float* __restrict__ A,
                                                   const float* __restrict__ W,
                                                   _Float16* __restrict__ C) {
    __shared__ _Float16 As[2][128][40];   // 20480 B
    __shared__ _Float16 Ws[2][64][40];    // 10240 B
    const int tid  = threadIdx.x;
    const int lane = tid & 31, wave = tid >> 5;
    const int row0 = blockIdx.x * 128, col0 = blockIdx.y * 64;
    const int rt = wave & 3, ct = wave >> 2;         // wave -> 32-row band, 32-col band
    const int m = lane & 15, half = lane >> 4;

    // staging: A tile 128x32 = 512 groups of 8 (2/thread), W tile 64x32 = 256 groups
    const int ar0 = tid >> 2, ar1 = 64 + (tid >> 2);
    const int acg = (tid & 3) * 8;
    const float* pa0 = A + (size_t)(row0 + ar0) * H_DIM + acg;
    const float* pa1 = A + (size_t)(row0 + ar1) * H_DIM + acg;
    const float* pw  = W + (size_t)(col0 + ar0) * H_DIM + acg;

    v8f acc00 = {0.f,0.f,0.f,0.f,0.f,0.f,0.f,0.f};
    v8f acc01 = acc00, acc10 = acc00, acc11 = acc00;

    // prologue: tile 0 -> buffer 0
    {
        float4 x0 = *(const float4*)(pa0),     x1 = *(const float4*)(pa0 + 4);
        float4 y0 = *(const float4*)(pa1),     y1 = *(const float4*)(pa1 + 4);
        float4 z0 = *(const float4*)(pw),      z1 = *(const float4*)(pw + 4);
        Pack8 t0, t1, t2; cvt8(x0, x1, t0); cvt8(y0, y1, t1); cvt8(z0, z1, t2);
        *(uint4*)&As[0][ar0][acg] = t0.q;
        *(uint4*)&As[0][ar1][acg] = t1.q;
        *(uint4*)&Ws[0][ar0][acg] = t2.q;
    }

    const int NT = H_DIM / 32;   // 160
    int cur = 0;
    for (int i = 0; i < NT; ++i) {
        __syncthreads();
        const bool has_next = (i + 1) < NT;
        float4 x0, x1, y0, y1, z0, z1;
        if (has_next) {
            const int kn_ = (i + 1) * 32;
            x0 = *(const float4*)(pa0 + kn_);  x1 = *(const float4*)(pa0 + kn_ + 4);
            y0 = *(const float4*)(pa1 + kn_);  y1 = *(const float4*)(pa1 + kn_ + 4);
            z0 = *(const float4*)(pw  + kn_);  z1 = *(const float4*)(pw  + kn_ + 4);
        }
        FragH fa0, fa1, fb0, fb1;
        fa0.q[0] = *(const uint4*)&As[cur][rt*32 + m][8*half];
        fa0.q[1] = *(const uint4*)&As[cur][rt*32 + m][16 + 8*half];
        fa1.q[0] = *(const uint4*)&As[cur][rt*32 + 16 + m][8*half];
        fa1.q[1] = *(const uint4*)&As[cur][rt*32 + 16 + m][16 + 8*half];
        fb0.q[0] = *(const uint4*)&Ws[cur][ct*32 + m][8*half];
        fb0.q[1] = *(const uint4*)&Ws[cur][ct*32 + m][16 + 8*half];
        fb1.q[0] = *(const uint4*)&Ws[cur][ct*32 + 16 + m][8*half];
        fb1.q[1] = *(const uint4*)&Ws[cur][ct*32 + 16 + m][16 + 8*half];
        acc00 = wmma_f16(fa0, fb0, acc00);
        acc01 = wmma_f16(fa0, fb1, acc01);
        acc10 = wmma_f16(fa1, fb0, acc10);
        acc11 = wmma_f16(fa1, fb1, acc11);
        if (has_next) {
            Pack8 t0, t1, t2; cvt8(x0, x1, t0); cvt8(y0, y1, t1); cvt8(z0, z1, t2);
            const int nxt = cur ^ 1;
            *(uint4*)&As[nxt][ar0][acg] = t0.q;
            *(uint4*)&As[nxt][ar1][acg] = t1.q;
            *(uint4*)&Ws[nxt][ar0][acg] = t2.q;
        }
        cur ^= 1;
    }

    // C/D layout: lane holds rows v+8*half, col (lane&15)
    const int orow = row0 + rt * 32 + 8 * half;
    const int ocol = col0 + ct * 32 + m;
#pragma unroll
    for (int v = 0; v < 8; ++v) {
        C[(size_t)(orow + v)      * P_DIM + ocol]      = (_Float16)acc00[v];
        C[(size_t)(orow + v)      * P_DIM + ocol + 16] = (_Float16)acc01[v];
        C[(size_t)(orow + 16 + v) * P_DIM + ocol]      = (_Float16)acc10[v];
        C[(size_t)(orow + 16 + v) * P_DIM + ocol + 16] = (_Float16)acc11[v];
    }
}

// --------------------------------------------------------------------------
// Row squared norms of f16 [rows x P] (consistent with GEMM inputs).
// --------------------------------------------------------------------------
__global__ __launch_bounds__(256) void norm_kernel(const _Float16* __restrict__ X,
                                                   float* __restrict__ nrm) {
    const int row = blockIdx.x;
    const _Float16* p = X + (size_t)row * P_DIM;
    float s = 0.f;
    for (int c = threadIdx.x; c < P_DIM; c += 256) {
        float v = (float)p[c];
        s += v * v;
    }
#pragma unroll
    for (int off = 16; off > 0; off >>= 1) s += __shfl_xor(s, off, 32);
    __shared__ float red[8];
    if ((threadIdx.x & 31) == 0) red[threadIdx.x >> 5] = s;
    __syncthreads();
    if (threadIdx.x == 0) {
        float t = 0.f;
#pragma unroll
        for (int i = 0; i < 8; ++i) t += red[i];
        nrm[row] = t;
    }
}

// --------------------------------------------------------------------------
// Fused distance GEMM + top-3 + softmax + gather.
// Block = 32 query rows (halves L2 traffic on kp); 8 waves, each wave takes
// a PAIR of adjacent column tiles (2x2 register blocking -> 4 WMMA / K-step).
// Top-3 merged cross-lane via __shfl_xor (masks 1..8 stay inside each
// 16-lane half == the lanes sharing a C-row), so LDS merge stage is tiny.
// --------------------------------------------------------------------------
__global__ __launch_bounds__(256) void topk_kernel(const float* __restrict__ values,
                                                   const _Float16* __restrict__ qf,
                                                   const _Float16* __restrict__ kpf,
                                                   const float* __restrict__ qn,
                                                   const float* __restrict__ kn,
                                                   float* __restrict__ out) {
    __shared__ _Float16 qs[32][P_DIM];   // 81920 B (CDNA5: 320 KB/WGP available)
    __shared__ float cd[32][24];
    __shared__ int   ci[32][24];
    __shared__ float swt[32][3];
    __shared__ int   sidx[32][3];

    const int tid  = threadIdx.x;
    const int lane = tid & 31, wave = tid >> 5;
    const int row0 = blockIdx.x * 32;
    const int nn = lane & 15, half = lane >> 4;

    // Stage 32 query rows (f16): 32*160 = 5120 groups of 8 halfs, 20 iters.
#pragma unroll
    for (int it = 0; it < 20; ++it) {
        int g  = tid + it * 256;
        int r  = g / 160;
        int c8 = (g % 160) * 8;
        *(uint4*)&qs[r][c8] = *(const uint4*)(qf + (size_t)(row0 + r) * P_DIM + c8);
    }
    float qn_r[2][8];
#pragma unroll
    for (int rt = 0; rt < 2; ++rt)
#pragma unroll
        for (int v = 0; v < 8; ++v)
            qn_r[rt][v] = qn[row0 + rt * 16 + v + 8 * half];
    __syncthreads();

    float bd[16][3];
    int   bi[16][3];
#pragma unroll
    for (int li = 0; li < 16; ++li) {
        bd[li][0] = bd[li][1] = bd[li][2] = 3.4028235e38f;
        bi[li][0] = bi[li][1] = bi[li][2] = 0;
    }

    for (int jtp = wave; jtp < M_MEM / 32; jtp += 8) {   // 32 cols per step
        const int c0 = jtp * 32 + nn;        // lane's column in tile pair
        const int c1 = c0 + 16;
        const float kn0 = kn[c0], kn1 = kn[c1];
        const _Float16* bp0 = kpf + (size_t)c0 * P_DIM;
        const _Float16* bp1 = kpf + (size_t)c1 * P_DIM;
        if (jtp + 8 < M_MEM / 32)
            __builtin_prefetch(bp0 + (size_t)256 * P_DIM, 0, 1);

        v8f acc00 = {0.f,0.f,0.f,0.f,0.f,0.f,0.f,0.f};
        v8f acc01 = acc00, acc10 = acc00, acc11 = acc00;
        for (int kk = 0; kk < P_DIM; kk += 32) {
            FragH fa0, fa1, fb0, fb1;
            fa0.q[0] = *(const uint4*)&qs[nn][kk + 8*half];
            fa0.q[1] = *(const uint4*)&qs[nn][kk + 16 + 8*half];
            fa1.q[0] = *(const uint4*)&qs[16 + nn][kk + 8*half];
            fa1.q[1] = *(const uint4*)&qs[16 + nn][kk + 16 + 8*half];
            fb0.q[0] = *(const uint4*)(bp0 + kk + 8*half);
            fb0.q[1] = *(const uint4*)(bp0 + kk + 16 + 8*half);
            fb1.q[0] = *(const uint4*)(bp1 + kk + 8*half);
            fb1.q[1] = *(const uint4*)(bp1 + kk + 16 + 8*half);
            acc00 = wmma_f16(fa0, fb0, acc00);
            acc01 = wmma_f16(fa0, fb1, acc01);
            acc10 = wmma_f16(fa1, fb0, acc10);
            acc11 = wmma_f16(fa1, fb1, acc11);
        }
#pragma unroll
        for (int v = 0; v < 8; ++v) {
            ins3(qn_r[0][v] + kn0 - 2.0f * acc00[v], c0, bd[v],     bi[v]);
            ins3(qn_r[0][v] + kn1 - 2.0f * acc01[v], c1, bd[v],     bi[v]);
            ins3(qn_r[1][v] + kn0 - 2.0f * acc10[v], c0, bd[8 + v], bi[8 + v]);
            ins3(qn_r[1][v] + kn1 - 2.0f * acc11[v], c1, bd[8 + v], bi[8 + v]);
        }
    }

    // Cross-lane top-3 reduction over the 16 lanes of each half.
#pragma unroll
    for (int li = 0; li < 16; ++li) {
#pragma unroll
        for (int mask = 1; mask <= 8; mask <<= 1) {
            float p0 = __shfl_xor(bd[li][0], mask, 32);
            int   q0 = __shfl_xor(bi[li][0], mask, 32);
            float p1 = __shfl_xor(bd[li][1], mask, 32);
            int   q1 = __shfl_xor(bi[li][1], mask, 32);
            float p2 = __shfl_xor(bd[li][2], mask, 32);
            int   q2 = __shfl_xor(bi[li][2], mask, 32);
            ins3(p0, q0, bd[li], bi[li]);
            ins3(p1, q1, bd[li], bi[li]);
            ins3(p2, q2, bd[li], bi[li]);
        }
    }
    if (nn == 0) {   // lane 0 / lane 16 publish per-wave top-3
#pragma unroll
        for (int li = 0; li < 16; ++li) {
            int row = (li >> 3) * 16 + (li & 7) + 8 * half;
#pragma unroll
            for (int s = 0; s < 3; ++s) {
                cd[row][wave * 3 + s] = bd[li][s];
                ci[row][wave * 3 + s] = bi[li][s];
            }
        }
    }
    __syncthreads();

    if (tid < 32) {
        float fd[3] = {3.4028235e38f, 3.4028235e38f, 3.4028235e38f};
        int   fi[3] = {0, 0, 0};
#pragma unroll
        for (int j = 0; j < 24; ++j) ins3(cd[tid][j], ci[tid][j], fd, fi);
        float d0 = sqrtf(fmaxf(fd[0], 1e-12f));
        float d1 = sqrtf(fmaxf(fd[1], 1e-12f));
        float d2 = sqrtf(fmaxf(fd[2], 1e-12f));
        float e1 = __expf(d0 - d1);          // softmax over -dist, max = -d0
        float e2 = __expf(d0 - d2);
        float inv = 1.0f / (1.0f + e1 + e2);
        swt[tid][0] = inv; swt[tid][1] = e1 * inv; swt[tid][2] = e2 * inv;
        sidx[tid][0] = fi[0]; sidx[tid][1] = fi[1]; sidx[tid][2] = fi[2];
    }
    __syncthreads();

    {   // weighted gather: 32 rows x 128 cols = 4096 floats, 16/thread
        int r  = tid >> 3;
        int eb = (tid & 7) * 16;
        float w0 = swt[r][0], w1 = swt[r][1], w2 = swt[r][2];
        const float* p0 = values + (size_t)sidx[r][0] * E_DIM + eb;
        const float* p1 = values + (size_t)sidx[r][1] * E_DIM + eb;
        const float* p2 = values + (size_t)sidx[r][2] * E_DIM + eb;
        float* po = out + (size_t)(row0 + r) * E_DIM + eb;
#pragma unroll
        for (int i = 0; i < 16; ++i) po[i] = w0 * p0[i] + w1 * p1[i] + w2 * p2[i];
    }
}

// --------------------------------------------------------------------------
extern "C" void kernel_launch(void* const* d_in, const int* in_sizes, int n_in,
                              void* d_out, int out_size, void* d_ws, size_t ws_size,
                              hipStream_t stream) {
    (void)in_sizes; (void)n_in; (void)out_size; (void)ws_size;
    const float* x      = (const float*)d_in[0];   // [N, H]
    const float* keys   = (const float*)d_in[1];   // [M, H]
    const float* values = (const float*)d_in[2];   // [M, E]
    const float* w_key  = (const float*)d_in[3];   // [P, H]
    float* out = (float*)d_out;                    // [N, E]

    char* ws = (char*)d_ws;
    constexpr size_t QF_SZ   = (size_t)N_Q  * P_DIM * sizeof(_Float16);
    constexpr size_t KPF_SZ  = (size_t)M_MEM * P_DIM * sizeof(_Float16);
    _Float16* qf  = (_Float16*)(ws);
    _Float16* kpf = (_Float16*)(ws + QF_SZ);
    float*    qn  = (float*)(ws + QF_SZ + KPF_SZ);
    float*    kn  = (float*)(ws + QF_SZ + KPF_SZ + (size_t)N_Q * sizeof(float));

    proj_kernel<<<dim3(N_Q  / 128, P_DIM / 64), 256, 0, stream>>>(x,    w_key, qf);
    proj_kernel<<<dim3(M_MEM / 128, P_DIM / 64), 256, 0, stream>>>(keys, w_key, kpf);

    norm_kernel<<<N_Q,   256, 0, stream>>>(qf,  qn);
    norm_kernel<<<M_MEM, 256, 0, stream>>>(kpf, kn);

    topk_kernel<<<N_Q / 32, 256, 0, stream>>>(values, qf, kpf, qn, kn, out);
}
